// GRU_38422777430357
// MI455X (gfx1250) — matrix operand order repaired and checked by reading, not compiled, not run on hardware
//
#include <hip/hip_runtime.h>
#include <hip/hip_bf16.h>
#include <stdint.h>

typedef __attribute__((ext_vector_type(16))) __bf16 v16bf;
typedef __attribute__((ext_vector_type(8)))  __bf16 v8bf;
typedef __attribute__((ext_vector_type(8)))  float  v8f;

#define BB   8192
#define TT   28
#define FF   28
#define HH   128
#define G3   384      // 3*H
#define BT   (BB*TT)  // 229376

// Async global->LDS copy, 16B per lane. LDS dest offset from a VGPR; global src is SGPR base plus VGPR byte offset.
__device__ __forceinline__ void async_copy16(unsigned lds_off, const void* gbase, unsigned goff) {
    asm volatile("global_load_async_to_lds_b128 %0, %1, %2"
                 :: "v"(lds_off), "v"(goff), "s"(gbase)
                 : "memory");
}
__device__ __forceinline__ void async_wait0() {
    asm volatile("s_wait_asynccnt 0" ::: "memory");
}
__device__ __forceinline__ unsigned lds_addr(const void* p) {
    // Generic LDS pointers carry the AS3 byte offset in their low 32 bits.
    return (unsigned)(uintptr_t)p;
}

// ---------------------------------------------------------------------------
// Convert fp32 (rows x K) -> bf16 (rows x Kpad), zero-padding K..Kpad-1.
// ---------------------------------------------------------------------------
__global__ void cvt_pad_kernel(const float* __restrict__ src, __bf16* __restrict__ dst,
                               long rows, int K, int Kpad) {
    long i = (long)blockIdx.x * 256 + threadIdx.x;
    long total = rows * (long)Kpad;
    if (i >= total) return;
    long r = i / Kpad;
    int  c = (int)(i - r * Kpad);
    float v = (c < K) ? src[r * (long)K + c] : 0.0f;
    dst[i] = (__bf16)v;
}

// ---------------------------------------------------------------------------
// gi = A(bf16, M x KPAD) @ W(bf16, 384 x KPAD)^T + bias -> bf16 (M x 384)
// Block = 128 A-rows x full N=384. A tile + W staged in LDS via async copies;
// each of 8 waves computes 8 M-subtiles x 3 N-tiles = 24 C tiles (96 WMMAs).
// ---------------------------------------------------------------------------
template<int KPAD, int KCHUNKS>
__global__ __launch_bounds__(256)
void gi_gemm_kernel(const __bf16* __restrict__ A, const __bf16* __restrict__ W,
                    const float* __restrict__ bias, __bf16* __restrict__ out) {
    __shared__ __bf16 s_a[128][KPAD + 8];
    __shared__ __bf16 s_w[G3][KPAD + 8];

    const int tid = threadIdx.x;
    const long m0 = (long)blockIdx.x * 128;
    const int cpr = KPAD / 8;                 // 16B chunks per row

    {   // ---- async stage A tile (128 x KPAD) and W (384 x KPAD) ----
        const unsigned abase = lds_addr(&s_a[0][0]);
        const __bf16* ga = A + m0 * KPAD;
        for (int i = tid; i < 128 * cpr; i += 256) {
            int n = i / cpr, c = i - n * cpr;
            async_copy16(abase + (unsigned)(n * (KPAD + 8) + c * 8) * 2u, ga,
                         (unsigned)i * 16u);
        }
        const unsigned wbase = lds_addr(&s_w[0][0]);
        for (int i = tid; i < G3 * cpr; i += 256) {
            int n = i / cpr, c = i - n * cpr;
            async_copy16(wbase + (unsigned)(n * (KPAD + 8) + c * 8) * 2u, W,
                         (unsigned)i * 16u);
        }
    }
    async_wait0();
    __syncthreads();

    const int wave = tid >> 5;
    const int lane = tid & 31;
    const int row  = lane & 15;
    const int hi   = lane >> 4;               // 0: lanes 0-15, 1: lanes 16-31

    for (int ms = 0; ms < 8; ++ms) {
        #pragma unroll
        for (int nti = 0; nti < 3; ++nti) {
            const int nt = wave * 3 + nti;
            v8f acc = {};
            #pragma unroll
            for (int kc = 0; kc < KCHUNKS; ++kc) {
                int ka = kc * 32 + hi * 16;   // A frag: K = {ka..+15}
                v8bf a0 = *(const v8bf*)&s_a[ms * 16 + row][ka];
                v8bf a1 = *(const v8bf*)&s_a[ms * 16 + row][ka + 8];
                v16bf a;
                #pragma unroll
                for (int i = 0; i < 8; ++i) { a[i] = a0[i]; a[i + 8] = a1[i]; }
                int kb = kc * 32 + hi * 16;   // B frag: 16 consecutive K of W row
                v16bf b = *(const v16bf*)&s_w[nt * 16 + row][kb];
                acc = __builtin_amdgcn_wmma_f32_16x16x32_bf16(false, a, false, b,
                                                              (short)0, acc, false, false);
            }
            const int  ncol = nt * 16 + row;
            const float bv  = bias[ncol];
            long ob = (m0 + ms * 16 + hi * 8) * (long)G3 + ncol;
            #pragma unroll
            for (int r = 0; r < 8; ++r)
                out[ob + (long)r * G3] = (__bf16)(acc[r] + bv);
        }
    }
}

// ---------------------------------------------------------------------------
// GRU recurrent scan. One block = 16 batch rows; w_hh (bf16, pre-converted)
// async-staged into LDS once; h carried fp32 in LDS with bf16 shadow as the
// WMMA A operand. Per step: 12 WMMAs/wave for gh = h @ w_hh^T, fused gates.
// ---------------------------------------------------------------------------
__global__ __launch_bounds__(256)
void gru_scan_kernel(const __bf16* __restrict__ gi, const __bf16* __restrict__ whh_bf,
                     const float* __restrict__ bhh, __bf16* __restrict__ hout) {
    __shared__ __bf16 s_w[G3][HH + 8];    // 104,448 B (row stride 272 B)
    __shared__ float  s_gh[16][G3 + 8];   //  25,088 B
    __shared__ float  s_h[16][HH + 4];    //   8,448 B
    __shared__ __bf16 s_hb[16][HH + 8];   //   4,352 B
    __shared__ float  s_b[G3];            //   1,536 B -> 143,872 B total

    const int tid = threadIdx.x;

    {   // ---- async stage w_hh (384x128 bf16 = 96 KB) into padded LDS ----
        const unsigned wbase = lds_addr(&s_w[0][0]);
        for (int i = tid; i < G3 * (HH / 8); i += 256) {   // 6144 16B chunks
            int n = i >> 4, c = i & 15;
            async_copy16(wbase + (unsigned)(n * (HH + 8) + c * 8) * 2u, whh_bf,
                         (unsigned)i * 16u);
        }
    }
    for (int i = tid; i < G3; i += 256) s_b[i] = bhh[i];
    for (int i = tid; i < 16 * HH; i += 256) {
        int r = i >> 7, c = i & (HH - 1);
        s_h[r][c]  = 0.0f;
        s_hb[r][c] = (__bf16)0.0f;
    }
    async_wait0();
    __syncthreads();

    const int wave = tid >> 5;
    const int lane = tid & 31;
    const int row  = lane & 15;
    const int hi   = lane >> 4;
    const long m0  = (long)blockIdx.x * 16;

    for (int t = 0; t < TT; ++t) {
        // ---- gh = h @ w_hh^T : each wave owns 3 N-tiles of 16 ----
        #pragma unroll
        for (int nti = 0; nti < 3; ++nti) {
            const int nt = wave * 3 + nti;
            v8f acc = {};
            #pragma unroll
            for (int kc = 0; kc < 4; ++kc) {
                int ka = kc * 32 + hi * 16;
                v8bf a0 = *(const v8bf*)&s_hb[row][ka];
                v8bf a1 = *(const v8bf*)&s_hb[row][ka + 8];
                v16bf a;
                #pragma unroll
                for (int i = 0; i < 8; ++i) { a[i] = a0[i]; a[i + 8] = a1[i]; }
                int kb = kc * 32 + hi * 16;
                v16bf b = *(const v16bf*)&s_w[nt * 16 + row][kb];
                acc = __builtin_amdgcn_wmma_f32_16x16x32_bf16(false, a, false, b,
                                                              (short)0, acc, false, false);
            }
            #pragma unroll
            for (int r = 0; r < 8; ++r)
                s_gh[hi * 8 + r][nt * 16 + row] = acc[r];
        }
        __syncthreads();

        // ---- fused gates: 16*128 elements, 8 per thread ----
        #pragma unroll
        for (int e = 0; e < 8; ++e) {
            int flat = tid + e * 256;
            int r = flat >> 7;
            int j = flat & (HH - 1);
            const __bf16* g = gi + ((m0 + r) * TT + t) * (long)G3;
            float gir = (float)g[j];
            float giz = (float)g[HH + j];
            float gin = (float)g[2 * HH + j];
            float ghr = s_gh[r][j]          + s_b[j];
            float ghz = s_gh[r][HH + j]     + s_b[HH + j];
            float ghn = s_gh[r][2 * HH + j] + s_b[2 * HH + j];
            float rr = 1.0f / (1.0f + __expf(-(gir + ghr)));
            float zz = 1.0f / (1.0f + __expf(-(giz + ghz)));
            float nn = tanhf(gin + rr * ghn);
            float hnew = (1.0f - zz) * nn + zz * s_h[r][j];
            s_h[r][j]  = hnew;
            s_hb[r][j] = (__bf16)hnew;
            hout[((m0 + r) * TT + t) * (long)HH + j] = (__bf16)hnew;
        }
        __syncthreads();
    }
}

// ---------------------------------------------------------------------------
// Head: out = relu(last @ w1^T + b1) @ w2^T + b2. One batch row per thread;
// weight accesses are thread-uniform -> scalar loads + broadcast.
// ---------------------------------------------------------------------------
__global__ __launch_bounds__(256)
void head_kernel(const __bf16* __restrict__ hseq, const float* __restrict__ w1,
                 const float* __restrict__ b1, const float* __restrict__ w2,
                 const float* __restrict__ b2, float* __restrict__ out) {
    int r = blockIdx.x * 256 + threadIdx.x;
    if (r >= BB) return;
    const __bf16* hp = hseq + ((long)r * TT + (TT - 1)) * HH;

    float hid[32];
    #pragma unroll
    for (int o = 0; o < 32; ++o) hid[o] = b1[o];
    for (int k = 0; k < HH; ++k) {
        float lv = (float)hp[k];
        #pragma unroll
        for (int o = 0; o < 32; ++o) hid[o] += lv * w1[o * HH + k];
    }
    #pragma unroll
    for (int o = 0; o < 32; ++o) hid[o] = hid[o] > 0.0f ? hid[o] : 0.0f;

    float o10[10];
    #pragma unroll
    for (int o = 0; o < 10; ++o) o10[o] = b2[o];
    #pragma unroll
    for (int k = 0; k < 32; ++k) {
        float lv = hid[k];
        #pragma unroll
        for (int o = 0; o < 10; ++o) o10[o] += lv * w2[o * 32 + k];
    }
    #pragma unroll
    for (int o = 0; o < 10; ++o) out[(long)r * 10 + o] = o10[o];
}

// ---------------------------------------------------------------------------
extern "C" void kernel_launch(void* const* d_in, const int* in_sizes, int n_in,
                              void* d_out, int out_size, void* d_ws, size_t ws_size,
                              hipStream_t stream) {
    const float* x     = (const float*)d_in[0];
    const float* w_ih0 = (const float*)d_in[1];
    const float* w_hh0 = (const float*)d_in[2];
    const float* b_ih0 = (const float*)d_in[3];
    const float* b_hh0 = (const float*)d_in[4];
    const float* w_ih1 = (const float*)d_in[5];
    const float* w_hh1 = (const float*)d_in[6];
    const float* b_ih1 = (const float*)d_in[7];
    const float* b_hh1 = (const float*)d_in[8];
    const float* w_ih2 = (const float*)d_in[9];
    const float* w_hh2 = (const float*)d_in[10];
    const float* b_ih2 = (const float*)d_in[11];
    const float* b_hh2 = (const float*)d_in[12];
    const float* w1    = (const float*)d_in[13];
    const float* bb1   = (const float*)d_in[14];
    const float* w2    = (const float*)d_in[15];
    const float* bb2   = (const float*)d_in[16];
    float* out = (float*)d_out;

    // Workspace carve-up (~308 MB total).
    char* p = (char*)d_ws;
    __bf16* gi   = (__bf16*)p; p += (long)BT * G3 * 2;   // 176.2 MB
    __bf16* hA   = (__bf16*)p; p += (long)BT * HH * 2;   //  58.7 MB
    __bf16* hB   = (__bf16*)p; p += (long)BT * HH * 2;   //  58.7 MB
    __bf16* xp   = (__bf16*)p; p += (long)BT * 32 * 2;   //  14.7 MB
    __bf16* wb0  = (__bf16*)p; p += (long)G3 * 32 * 2;
    __bf16* wb1  = (__bf16*)p; p += (long)G3 * HH * 2;
    __bf16* wb2  = (__bf16*)p; p += (long)G3 * HH * 2;
    __bf16* whb0 = (__bf16*)p; p += (long)G3 * HH * 2;
    __bf16* whb1 = (__bf16*)p; p += (long)G3 * HH * 2;
    __bf16* whb2 = (__bf16*)p; p += (long)G3 * HH * 2;

    // --- convert inputs/weights to bf16 (pad K 28 -> 32 for layer 0) ---
    {
        long tot = (long)BT * 32;
        cvt_pad_kernel<<<(int)((tot + 255) / 256), 256, 0, stream>>>(x, xp, BT, FF, 32);
    }
    cvt_pad_kernel<<<(G3 * 32 + 255) / 256, 256, 0, stream>>>(w_ih0, wb0, G3, FF, 32);
    cvt_pad_kernel<<<(G3 * HH + 255) / 256, 256, 0, stream>>>(w_ih1, wb1, G3, HH, HH);
    cvt_pad_kernel<<<(G3 * HH + 255) / 256, 256, 0, stream>>>(w_ih2, wb2, G3, HH, HH);
    cvt_pad_kernel<<<(G3 * HH + 255) / 256, 256, 0, stream>>>(w_hh0, whb0, G3, HH, HH);
    cvt_pad_kernel<<<(G3 * HH + 255) / 256, 256, 0, stream>>>(w_hh1, whb1, G3, HH, HH);
    cvt_pad_kernel<<<(G3 * HH + 255) / 256, 256, 0, stream>>>(w_hh2, whb2, G3, HH, HH);

    const int gemm_blocks = BT / 128;   // 1792
    const int scan_blocks = BB / 16;    // 512

    // --- layer 0 ---
    gi_gemm_kernel<32, 1><<<gemm_blocks, 256, 0, stream>>>(xp, wb0, b_ih0, gi);
    gru_scan_kernel<<<scan_blocks, 256, 0, stream>>>(gi, whb0, b_hh0, hA);
    // --- layer 1 ---
    gi_gemm_kernel<128, 4><<<gemm_blocks, 256, 0, stream>>>(hA, wb1, b_ih1, gi);
    gru_scan_kernel<<<scan_blocks, 256, 0, stream>>>(gi, whb1, b_hh1, hB);
    // --- layer 2 ---
    gi_gemm_kernel<128, 4><<<gemm_blocks, 256, 0, stream>>>(hB, wb2, b_ih2, gi);
    gru_scan_kernel<<<scan_blocks, 256, 0, stream>>>(gi, whb2, b_hh2, hA);
    // --- head ---
    head_kernel<<<(BB + 255) / 256, 256, 0, stream>>>(hA, w1, bb1, w2, bb2, out);
}
